// RetaGNN_SA_Model_5514738008105
// MI455X (gfx1250) — matrix operand order repaired and checked by reading, not compiled, not run on hardware
//
#include <hip/hip_runtime.h>
#include <hip/hip_bf16.h>
#include <hip/hip_fp16.h>

#define B_    64
#define N_    2048
#define E_    8192
#define L_    50
#define LP_   64      // L padded to multiple of 16 for WMMA
#define D_    128
#define R_    5
#define BASES_ 4
#define QKVD_ 192     // HEADS*HID = 3*64
#define YW_   768     // (R + 1 root) * 128 columns
#define QKVW_ 576     // 3 * 192

typedef __attribute__((ext_vector_type(16))) _Float16 v16h;
typedef __attribute__((ext_vector_type(8)))  _Float16 v8h;
typedef __attribute__((ext_vector_type(8)))  float    v8f;

// ---------------------------------------------------------------------------
// K0: build transposed f16 weight matrices.
//   WcatT[col][i], col = r*128+o for r<5 (basis combo), col=640+o -> root_w
//   WqkvT[col][i], col in [0,576) -> wq|wk|wv columns
// ---------------------------------------------------------------------------
__global__ void prep_weights(const float* __restrict__ bases,
                             const float* __restrict__ comp,
                             const float* __restrict__ root_w,
                             const float* __restrict__ wq,
                             const float* __restrict__ wk,
                             const float* __restrict__ wv,
                             _Float16* __restrict__ WcatT,
                             _Float16* __restrict__ WqkvT) {
  int idx = blockIdx.x * blockDim.x + threadIdx.x;
  const int totA = YW_ * D_;
  const int totB = QKVW_ * D_;
  if (idx < totA) {
    int col = idx / D_;
    int i   = idx % D_;
    int r   = col >> 7;
    int o   = col & (D_ - 1);
    float v;
    if (r < R_) {
      v = 0.f;
      #pragma unroll
      for (int bb = 0; bb < BASES_; ++bb)
        v += comp[r * BASES_ + bb] * bases[((size_t)bb * D_ + i) * D_ + o];
    } else {
      v = root_w[(size_t)i * D_ + o];
    }
    WcatT[(size_t)col * D_ + i] = (_Float16)v;
  } else if (idx < totA + totB) {
    int j   = idx - totA;
    int col = j / D_;
    int i   = j % D_;
    float v;
    if (col < QKVD_)          v = wq[(size_t)i * QKVD_ + col];
    else if (col < 2 * QKVD_) v = wk[(size_t)i * QKVD_ + (col - QKVD_)];
    else                      v = wv[(size_t)i * QKVD_ + (col - 2 * QKVD_)];
    WqkvT[(size_t)col * D_ + i] = (_Float16)v;
  }
}

// ---------------------------------------------------------------------------
// K1: x16[b][n][d] = f16(emb_table[node_ids[b][n]][d])
// ---------------------------------------------------------------------------
__global__ void gather_x(const float* __restrict__ emb,
                         const int* __restrict__ node_ids,
                         _Float16* __restrict__ x16) {
  size_t idx = (size_t)blockIdx.x * blockDim.x + threadIdx.x;
  if (idx >= (size_t)B_ * N_ * D_) return;
  int d = (int)(idx & (D_ - 1));
  size_t bn = idx >> 7;
  int nid = node_ids[bn];
  x16[idx] = (_Float16)emb[(size_t)nid * D_ + d];
}

// ---------------------------------------------------------------------------
// K2: batched WMMA GEMM.  C[b] = A[b] (MxK f16) * BT^T (BT is [N][K] f16)
// One wave (32 lanes) computes a 16(M) x 64(N) tile: 1 A fragment feeds 4
// v_wmma_f32_16x16x32_f16 per K-step. Fragment packing (this code's contiguous K order, same for A and B):
//   A: lanes 0-15 = rows, hold K 0..15; lanes 16-31 hold K 16..31
//   B: lanes 0-15 = cols, hold K 0..15; lanes 16-31 hold K 16..31
//   C/D: vgpr v, lane l -> (M = (l>>4)*8 + v, N = l&15)
// ---------------------------------------------------------------------------
__global__ void wmma_gemm_f16f32(const _Float16* __restrict__ A,
                                 const _Float16* __restrict__ BT,
                                 float* __restrict__ C,
                                 int M, int N, int K) {
  const int lane = threadIdx.x;       // 0..31
  const int hi   = lane >> 4;         // 0/1
  const int lr   = lane & 15;
  const int bz   = blockIdx.z;
  const int m0   = blockIdx.y * 16;
  const int n0   = blockIdx.x * 64;

  const _Float16* Arow = A + ((size_t)bz * M + m0 + lr) * K;

  v8f acc0 = {}, acc1 = {}, acc2 = {}, acc3 = {};

  for (int k0 = 0; k0 < K; k0 += 32) {
    v16h a = *(const v16h*)(Arow + k0 + hi * 16);

    const _Float16* Bbase = BT + (size_t)(n0 + lr) * K + k0 + hi * 16;
    v16h b0 = *(const v16h*)(Bbase + (size_t)0 * 16 * K);
    v16h b1 = *(const v16h*)(Bbase + (size_t)1 * 16 * K);
    v16h b2 = *(const v16h*)(Bbase + (size_t)2 * 16 * K);
    v16h b3 = *(const v16h*)(Bbase + (size_t)3 * 16 * K);

    acc0 = __builtin_amdgcn_wmma_f32_16x16x32_f16(false, a, false, b0, (short)0, acc0, false, false);
    acc1 = __builtin_amdgcn_wmma_f32_16x16x32_f16(false, a, false, b1, (short)0, acc1, false, false);
    acc2 = __builtin_amdgcn_wmma_f32_16x16x32_f16(false, a, false, b2, (short)0, acc2, false, false);
    acc3 = __builtin_amdgcn_wmma_f32_16x16x32_f16(false, a, false, b3, (short)0, acc3, false, false);
  }

  float* Crow = C + ((size_t)bz * M + m0 + hi * 8) * N + n0 + lr;
  #pragma unroll
  for (int v = 0; v < 8; ++v) {
    Crow[(size_t)v * N +  0] = acc0[v];
    Crow[(size_t)v * N + 16] = acc1[v];
    Crow[(size_t)v * N + 32] = acc2[v];
    Crow[(size_t)v * N + 48] = acc3[v];
  }
}

// ---------------------------------------------------------------------------
// K3: Y[b][n][640+d] += root_b[d]   (root block becomes x_upd accumulator)
// ---------------------------------------------------------------------------
__global__ void add_root_bias(float* __restrict__ Y, const float* __restrict__ root_b) {
  size_t idx = (size_t)blockIdx.x * blockDim.x + threadIdx.x;
  if (idx >= (size_t)B_ * N_ * D_) return;
  int d = (int)(idx & (D_ - 1));
  size_t bn = idx >> 7;
  Y[bn * YW_ + 5 * D_ + d] += root_b[d];
}

// ---------------------------------------------------------------------------
// K4/K5: per-(tgt, relation) degree counting (zero every launch: replay-safe)
// ---------------------------------------------------------------------------
__global__ void zero_deg(float* __restrict__ deg, size_t n) {
  size_t idx = (size_t)blockIdx.x * blockDim.x + threadIdx.x;
  if (idx < n) deg[idx] = 0.f;
}

__global__ void count_deg(const int* __restrict__ edge_index,
                          const int* __restrict__ edge_type,
                          float* __restrict__ deg) {
  size_t idx = (size_t)blockIdx.x * blockDim.x + threadIdx.x;
  if (idx >= (size_t)B_ * E_) return;
  int b = (int)(idx / E_);
  int e = (int)(idx % E_);
  int tgt = edge_index[(size_t)b * 2 * E_ + E_ + e];
  int et  = edge_type[(size_t)b * E_ + e];
  atomicAdd(&deg[((size_t)b * N_ + tgt) * R_ + et], 1.0f);
}

// ---------------------------------------------------------------------------
// K6: scatter normalized messages into root block of Y.
//   msg[e][d] = Y[b][src][et*128 + d]  (relation columns: read-only here)
// ---------------------------------------------------------------------------
__global__ void scatter_msgs(const int* __restrict__ edge_index,
                             const int* __restrict__ edge_type,
                             const float* __restrict__ deg,
                             float* __restrict__ Y) {
  size_t idx = (size_t)blockIdx.x * blockDim.x + threadIdx.x;
  if (idx >= (size_t)B_ * E_ * D_) return;
  int d = (int)(idx & (D_ - 1));
  size_t eg = idx >> 7;
  int b = (int)(eg / E_);
  int e = (int)(eg % E_);
  const int* ei = edge_index + (size_t)b * 2 * E_;
  int src = ei[e];
  int tgt = ei[E_ + e];
  int et  = edge_type[(size_t)b * E_ + e];
  float dg   = deg[((size_t)b * N_ + tgt) * R_ + et];
  float norm = 1.0f / fmaxf(dg, 1.0f);
  float msg  = Y[((size_t)b * N_ + src) * YW_ + (size_t)et * D_ + d];
  atomicAdd(&Y[((size_t)b * N_ + tgt) * YW_ + 5 * D_ + d], msg * norm);
}

// ---------------------------------------------------------------------------
// K7: X16[b][l][d] = f16(x_upd[b][sent_ids[b][l]][d]), rows L_..LP_-1 zeroed
// ---------------------------------------------------------------------------
__global__ void gather_sent(const float* __restrict__ Y,
                            const int* __restrict__ sent_ids,
                            _Float16* __restrict__ X16) {
  size_t idx = (size_t)blockIdx.x * blockDim.x + threadIdx.x;
  if (idx >= (size_t)B_ * LP_ * D_) return;
  int d   = (int)(idx & (D_ - 1));
  int l   = (int)((idx >> 7) & (LP_ - 1));
  int b   = (int)(idx >> 13);          // /(LP_*D_)
  if (l < L_) {
    int n = sent_ids[(size_t)b * L_ + l];
    X16[idx] = (_Float16)Y[((size_t)b * N_ + n) * YW_ + 5 * D_ + d];
  } else {
    X16[idx] = (_Float16)0.f;
  }
}

// ---------------------------------------------------------------------------
// K8: one block per (b, i): softmax(QK^T/sqrt(192)) row + H row = scores @ V
//   QKV layout [B][LP_][576]: cols 0..191=Q, 192..383=K, 384..575=V
// ---------------------------------------------------------------------------
__global__ void attn_kernel(const float* __restrict__ QKV, float* __restrict__ H) {
  int bi = blockIdx.x;
  int b  = bi / L_;
  int i  = bi % L_;
  const float* base = QKV + (size_t)b * LP_ * QKVW_;
  __shared__ float s[L_];
  __shared__ float invsum;
  int t = threadIdx.x;

  if (t < L_) {
    const float* qi = base + (size_t)i * QKVW_;
    const float* kj = base + (size_t)t * QKVW_ + QKVD_;
    float dot = 0.f;
    for (int d = 0; d < QKVD_; ++d) dot += qi[d] * kj[d];
    s[t] = dot * 0.07216878364870323f;   // 1/sqrt(192)
  }
  __syncthreads();
  if (t == 0) {
    float mx = s[0];
    for (int j = 1; j < L_; ++j) mx = fmaxf(mx, s[j]);
    float sum = 0.f;
    for (int j = 0; j < L_; ++j) { s[j] = __expf(s[j] - mx); sum += s[j]; }
    invsum = 1.f / sum;
  }
  __syncthreads();
  if (t < QKVD_) {
    float h = 0.f;
    for (int j = 0; j < L_; ++j)
      h += s[j] * base[(size_t)j * QKVW_ + 2 * QKVD_ + t];
    H[((size_t)b * L_ + i) * QKVD_ + t] = h * invsum;
  }
}

// ---------------------------------------------------------------------------
// K9: per-batch attention pooling + fc + sigmoid. Block = 256 threads.
// ---------------------------------------------------------------------------
__global__ void pool_kernel(const float* __restrict__ H,
                            const float* __restrict__ attn_w,
                            const float* __restrict__ attn_b,
                            const float* __restrict__ fc_w,
                            const float* __restrict__ fc_b,
                            float* __restrict__ out) {
  int b = blockIdx.x;
  int t = threadIdx.x;
  __shared__ float red[256];
  __shared__ float a_l;
  const float* Hb = H + (size_t)b * L_ * QKVD_;
  float pooled = 0.f;
  for (int l = 0; l < L_; ++l) {
    float hv = (t < QKVD_) ? Hb[(size_t)l * QKVD_ + t] : 0.f;
    red[t] = (t < QKVD_) ? hv * attn_w[t] : 0.f;
    __syncthreads();
    for (int st = 128; st > 0; st >>= 1) {
      if (t < st) red[t] += red[t + st];
      __syncthreads();
    }
    if (t == 0) a_l = red[0] + attn_b[0];
    __syncthreads();
    pooled += hv * a_l;
    __syncthreads();
  }
  red[t] = (t < QKVD_) ? pooled * fc_w[t] : 0.f;
  __syncthreads();
  for (int st = 128; st > 0; st >>= 1) {
    if (t < st) red[t] += red[t + st];
    __syncthreads();
  }
  if (t == 0) out[b] = 1.f / (1.f + __expf(-(red[0] + fc_b[0])));
}

// ---------------------------------------------------------------------------
extern "C" void kernel_launch(void* const* d_in, const int* in_sizes, int n_in,
                              void* d_out, int out_size, void* d_ws, size_t ws_size,
                              hipStream_t stream) {
  (void)in_sizes; (void)n_in; (void)out_size; (void)ws_size;
  const float* emb       = (const float*)d_in[0];
  const float* bases     = (const float*)d_in[1];
  const float* comp      = (const float*)d_in[2];
  const float* root_w    = (const float*)d_in[3];
  const float* root_b    = (const float*)d_in[4];
  const float* wq        = (const float*)d_in[5];
  const float* wk        = (const float*)d_in[6];
  const float* wv        = (const float*)d_in[7];
  const float* attn_w    = (const float*)d_in[8];
  const float* attn_b    = (const float*)d_in[9];
  const float* fc_w      = (const float*)d_in[10];
  const float* fc_b      = (const float*)d_in[11];
  const int* node_ids    = (const int*)d_in[12];
  const int* edge_index  = (const int*)d_in[13];
  const int* edge_type   = (const int*)d_in[14];
  const int* sent_ids    = (const int*)d_in[15];
  float* out             = (float*)d_out;

  char* ws = (char*)d_ws;
  size_t off = 0;
  auto alloc = [&](size_t bytes) -> char* {
    off = (off + 255) & ~(size_t)255;
    char* p = ws + off;
    off += bytes;
    return p;
  };
  _Float16* WcatT = (_Float16*)alloc((size_t)YW_ * D_ * sizeof(_Float16));
  _Float16* WqkvT = (_Float16*)alloc((size_t)QKVW_ * D_ * sizeof(_Float16));
  _Float16* x16   = (_Float16*)alloc((size_t)B_ * N_ * D_ * sizeof(_Float16));
  float*    Y     = (float*)   alloc((size_t)B_ * N_ * YW_ * sizeof(float));
  float*    deg   = (float*)   alloc((size_t)B_ * N_ * R_ * sizeof(float));
  _Float16* X16   = (_Float16*)alloc((size_t)B_ * LP_ * D_ * sizeof(_Float16));
  float*    QKV   = (float*)   alloc((size_t)B_ * LP_ * QKVW_ * sizeof(float));
  float*    Hbuf  = (float*)   alloc((size_t)B_ * L_ * QKVD_ * sizeof(float));

  {
    int tot = YW_ * D_ + QKVW_ * D_;
    prep_weights<<<(tot + 255) / 256, 256, 0, stream>>>(bases, comp, root_w, wq, wk, wv, WcatT, WqkvT);
  }
  {
    size_t tot = (size_t)B_ * N_ * D_;
    gather_x<<<(unsigned)((tot + 255) / 256), 256, 0, stream>>>(emb, node_ids, x16);
  }
  {
    dim3 g(YW_ / 64, N_ / 16, B_);
    wmma_gemm_f16f32<<<g, 32, 0, stream>>>(x16, WcatT, Y, N_, YW_, D_);
  }
  {
    size_t tot = (size_t)B_ * N_ * D_;
    add_root_bias<<<(unsigned)((tot + 255) / 256), 256, 0, stream>>>(Y, root_b);
  }
  {
    size_t tot = (size_t)B_ * N_ * R_;
    zero_deg<<<(unsigned)((tot + 255) / 256), 256, 0, stream>>>(deg, tot);
  }
  {
    size_t tot = (size_t)B_ * E_;
    count_deg<<<(unsigned)((tot + 255) / 256), 256, 0, stream>>>(edge_index, edge_type, deg);
  }
  {
    size_t tot = (size_t)B_ * E_ * D_;
    scatter_msgs<<<(unsigned)((tot + 255) / 256), 256, 0, stream>>>(edge_index, edge_type, deg, Y);
  }
  {
    size_t tot = (size_t)B_ * LP_ * D_;
    gather_sent<<<(unsigned)((tot + 255) / 256), 256, 0, stream>>>(Y, sent_ids, X16);
  }
  {
    dim3 g(QKVW_ / 64, LP_ / 16, B_);
    wmma_gemm_f16f32<<<g, 32, 0, stream>>>(X16, WqkvT, QKV, LP_, QKVW_, D_);
  }
  attn_kernel<<<B_ * L_, 256, 0, stream>>>(QKV, Hbuf);
  pool_kernel<<<B_, 256, 0, stream>>>(Hbuf, attn_w, attn_b, fc_w, fc_b, out);
}
